// PCNO_2250562863748
// MI455X (gfx1250) — matrix operand order repaired and checked by reading, not compiled, not run on hardware
//
#include <hip/hip_runtime.h>

typedef __bf16 bf16_t;
typedef __attribute__((ext_vector_type(16))) __bf16 v16bf;
typedef __attribute__((ext_vector_type(8)))  __bf16 v8bf;
typedef __attribute__((ext_vector_type(8)))  float  v8f;
typedef int v4i __attribute__((vector_size(16)));

#define B_   2
#define N_   16384
#define E_   262144
#define K_   256
#define D_   3
#define C_   128
#define L_   4
#define FC_  128
#define K2_  512      // 2K (cos|sin concatenated)
#define GD_  384      // C*D gradient channels

// ---- CDNA5 async global->LDS path (guarded; falls back to sync staging) ----
#if defined(__gfx1250__) && __has_builtin(__builtin_amdgcn_global_load_async_to_lds_b128) && \
    __has_builtin(__builtin_amdgcn_s_wait_asynccnt)
#define HAVE_ASYNC_LDS 1
typedef __attribute__((address_space(1))) v4i* gptr_v4i;
typedef __attribute__((address_space(3))) v4i* lptr_v4i;
#endif

static __device__ __forceinline__ void copy16_to_lds(bf16_t* ldsdst, const bf16_t* gsrc) {
#ifdef HAVE_ASYNC_LDS
    __builtin_amdgcn_global_load_async_to_lds_b128(
        (gptr_v4i)(bf16_t*)gsrc, (lptr_v4i)ldsdst, 0, 0);
#else
    *(v8bf*)ldsdst = *(const v8bf*)gsrc;
#endif
}
static __device__ __forceinline__ void stage_wait() {
#ifdef HAVE_ASYNC_LDS
    __builtin_amdgcn_s_wait_asynccnt(0);
#endif
    __syncthreads();
}

static __device__ __forceinline__ float gelu_exact(float x) {
    return 0.5f * x * (1.0f + erff(x * 0.70710678118654752440f));
}

// ---- WMMA bf16 fragment loaders (contiguous k order k = 16h + s, same for A and B) ----
// A 16x32 bf16, row-major source. rowk points at element (m, k_base) of A.
// lane<16 : elems 0..15 = K[0..15]
// lane>=16: elems 0..15 = K[16..31]
static __device__ __forceinline__ v16bf load_frag_a(const bf16_t* __restrict__ rowk, int lh) {
    const v8bf x0 = *(const v8bf*)(rowk + lh * 16);
    const v8bf x1 = *(const v8bf*)(rowk + lh * 16 + 8);
    v16bf a;
#pragma unroll
    for (int i = 0; i < 8; ++i) { a[i] = x0[i]; a[8 + i] = x1[i]; }
    return a;
}
// B 32x16 bf16 from B^T rows: lane holds column (l&15), 16 consecutive K
// starting at k + 16*(l>>4). p points at that element (global or LDS).
static __device__ __forceinline__ v16bf load_frag_b(const bf16_t* p) {
    const v8bf x0 = *(const v8bf*)(p);
    const v8bf x1 = *(const v8bf*)(p + 8);
    v16bf b;
#pragma unroll
    for (int i = 0; i < 8; ++i) { b[i] = x0[i]; b[8 + i] = x1[i]; }
    return b;
}

// ---- Fourier bases: write weighted [2K,N] (fwd B^T) and plain [N,2K] (inv B^T)
__global__ void bases_kernel(const float* __restrict__ nodes,
                             const float* __restrict__ node_mask,
                             const float* __restrict__ node_weights,
                             const float* __restrict__ modes,
                             const float* __restrict__ spL,
                             bf16_t* __restrict__ BasNT,   // [B][N][512]
                             bf16_t* __restrict__ WBTw) {  // [B][512][N]
    int t = blockIdx.x * blockDim.x + threadIdx.x;   // over B*N*K
    int k = t & (K_ - 1);
    int n = (t >> 8) & (N_ - 1);
    int b = t >> 22;
    if (b >= B_) return;
    const float* nd = nodes + (long)(b * N_ + n) * D_;
    float tmp = 0.f;
#pragma unroll
    for (int d = 0; d < D_; ++d) tmp += nd[d] * modes[k * D_ + d] * spL[d];
    float bc = cosf(tmp), bs = sinf(tmp);
    float nw = node_weights[b * N_ + n] * node_mask[b * N_ + n];
    bf16_t* bn = BasNT + ((long)b * N_ + n) * K2_;
    bn[k]      = (bf16_t)bc;
    bn[K_ + k] = (bf16_t)bs;
    bf16_t* wb = WBTw + (long)b * K2_ * N_;
    wb[(long)k * N_ + n]        = (bf16_t)(bc * nw);     // wbc
    wb[(long)(K_ + k) * N_ + n] = (bf16_t)(-bs * nw);    // -wbs  (xs = -h.wbs)
}

// ---- lift: h = fc0(x); write both layouts
__global__ void lift_kernel(const float* __restrict__ x,
                            const float* __restrict__ fc0_w,
                            const float* __restrict__ fc0_b,
                            bf16_t* __restrict__ h_bf,    // [B][C][N]
                            bf16_t* __restrict__ hcatT) { // [B][N][512] cols 0:128
    int bn = blockIdx.x;
    int c  = threadIdx.x;
    int b = bn >> 14, n = bn & (N_ - 1);
    const float* xp = x + (long)bn * 3;
    float acc = fc0_b[c];
#pragma unroll
    for (int i = 0; i < 3; ++i) acc += xp[i] * fc0_w[c * 3 + i];
    h_bf[((long)b * C_ + c) * N_ + n]      = (bf16_t)acc;
    hcatT[((long)b * N_ + n) * K2_ + c]    = (bf16_t)acc;
}

__global__ void zero_kernel(float* __restrict__ p, long count) {
    long i = (long)blockIdx.x * blockDim.x + threadIdx.x;
    long stride = (long)gridDim.x * blockDim.x;
    for (; i < count; i += stride) p[i] = 0.f;
}

__global__ void convert_kernel(const float* __restrict__ s, bf16_t* __restrict__ d, long n) {
    long i = (long)blockIdx.x * blockDim.x + threadIdx.x;
    long stride = (long)gridDim.x * blockDim.x;
    for (; i < n; i += stride) d[i] = (bf16_t)s[i];
}

// ---- edge gradient: one block per edge, thread = channel, 3 f32 atomics
__global__ void edge_grad_kernel(const int* __restrict__ edges,
                                 const float* __restrict__ egw,
                                 const bf16_t* __restrict__ hcatT, // [B][N][512] cols 0:128
                                 float* __restrict__ gradT) {      // [B][N][384]
    long be = blockIdx.x;       // b*E + e
    int c = threadIdx.x;
    long b = be >> 18;          // E = 2^18
    int tgt = edges[be * 2 + 0];
    int src = edges[be * 2 + 1];
    const bf16_t* hb = hcatT + b * (long)N_ * K2_;
    float diff = (float)hb[(long)src * K2_ + c] - (float)hb[(long)tgt * K2_ + c];
    float* g = gradT + (b * (long)N_ + tgt) * GD_ + c * 3;
    const float* w = egw + be * 3;
#pragma unroll
    for (int d = 0; d < 3; ++d) atomicAdd(&g[d], w[d] * diff);
}

__global__ void grad_convert_kernel(const float* __restrict__ gradT,
                                    bf16_t* __restrict__ hcatT) { // cols 128:512
    long i = (long)blockIdx.x * blockDim.x + threadIdx.x;
    long total = (long)B_ * N_ * GD_;
    long stride = (long)gridDim.x * blockDim.x;
    for (; i < total; i += stride) {
        long bn = i / GD_;
        int g = (int)(i - bn * GD_);
        hcatT[bn * K2_ + C_ + g] = (bf16_t)gradT[i];
    }
}

// ---- forward transform GEMM (split-K, f32 atomic reduce): Xf[b,128,512] = h x WBTw^T
__global__ __launch_bounds__(256) void gemm_fwd_kernel(const bf16_t* __restrict__ Abase, // [B][128][16384]
                                                       const bf16_t* __restrict__ BTbase,// [B][512][16384]
                                                       float* __restrict__ outF) {       // [B][128][512]
    int wave = threadIdx.x >> 5, lane = threadIdx.x & 31;
    int l15 = lane & 15, lh = lane >> 4;
    int b     = blockIdx.z >> 3;      // KSPLIT = 8
    int chunk = blockIdx.z & 7;
    int k0 = chunk * (N_ / 8);
    int mbase = wave * 16;
    int nbase = blockIdx.x * 64;
    const bf16_t* A  = Abase  + (long)b * C_ * N_ + (long)(mbase + l15) * N_;
    const bf16_t* BT = BTbase + (long)b * K2_ * N_;
    v8f acc[4] = {};
    for (int k = k0; k < k0 + N_ / 8; k += 32) {
        v16bf a = load_frag_a(A + k, lh);
#pragma unroll
        for (int j = 0; j < 4; ++j) {
            v16bf bb = load_frag_b(BT + (long)(nbase + j * 16 + l15) * N_ + k + lh * 16);
            acc[j] = __builtin_amdgcn_wmma_f32_16x16x32_bf16(false, a, false, bb,
                                                             (short)0, acc[j], false, false);
        }
    }
    float* o = outF + (long)b * C_ * K2_;
#pragma unroll
    for (int j = 0; j < 4; ++j) {
        int ncol = nbase + j * 16 + l15;
#pragma unroll
        for (int r = 0; r < 8; ++r) {
            int m = mbase + r + 8 * lh;
            atomicAdd(&o[(long)m * K2_ + ncol], acc[j][r]);
        }
    }
}

// ---- per-mode CxC mixing: G = [2fc | -2fs] bf16
__global__ void modemix_kernel(const float* __restrict__ Xf,  // [B][128][512]
                               const float* __restrict__ wc,  // layer slice [i][o][k]
                               const float* __restrict__ wsn,
                               bf16_t* __restrict__ G) {      // [B][128][512]
    int k = blockIdx.x;    // mode
    int o = threadIdx.x;   // out channel
    __shared__ float sxc[B_][C_], sxs[B_][C_];
    for (int b = 0; b < B_; ++b) {
        sxc[b][o] = Xf[((long)b * C_ + o) * K2_ + k];
        sxs[b][o] = Xf[((long)b * C_ + o) * K2_ + K_ + k];
    }
    __syncthreads();
    float fc[B_] = {0.f, 0.f}, fs[B_] = {0.f, 0.f};
    for (int i = 0; i < C_; ++i) {
        float wcv = wc[(i * C_ + o) * K_ + k];
        float wsv = wsn[(i * C_ + o) * K_ + k];
#pragma unroll
        for (int b = 0; b < B_; ++b) {
            float xc = sxc[b][i], xs = sxs[b][i];
            fc[b] += xc * wcv - xs * wsv;
            fs[b] += xs * wcv + xc * wsv;
        }
    }
    for (int b = 0; b < B_; ++b) {
        G[((long)b * C_ + o) * K2_ + k]      = (bf16_t)( 2.f * fc[b]);
        G[((long)b * C_ + o) * K2_ + K_ + k] = (bf16_t)(-2.f * fs[b]);
    }
}

// ---- build concatenated channel-mix weights [ws_w | gws_w] (bf16) + bias
__global__ void wcat_kernel(const float* __restrict__ ws_w, const float* __restrict__ gws_w,
                            const float* __restrict__ ws_b, const float* __restrict__ gws_b,
                            bf16_t* __restrict__ Wcat, float* __restrict__ biasCat) {
    int t = blockIdx.x * blockDim.x + threadIdx.x;
    if (t >= C_ * K2_) return;
    int o = t >> 9, g = t & 511;
    float v = (g < C_) ? ws_w[o * C_ + g] : gws_w[o * GD_ + (g - C_)];
    Wcat[t] = (bf16_t)v;
    if (g == 0) biasCat[o] = ws_b[o] + gws_b[o];
}

// ---- dual-reduction WMMA GEMM with async-LDS double-buffered B staging ----
// Out = A1 x B1T^T (+ A2 x B2T^T) + bias (+gelu); B tiles shared by all 8 waves
// are staged once per block into LDS (async DMA, ASYNCcnt) and re-read via DS.
#define ROWS_ 64          // n-columns per block
#define CK_   64          // K elements per stage chunk
#define LROW_ 72          // padded LDS row stride (144B) -> conflict-free banks

__global__ __launch_bounds__(256) void gemm_dual_kernel(
    const bf16_t* __restrict__ A1, long a1bs, int lda1, int K1,
    const bf16_t* __restrict__ B1T, long b1bs, int ldb1,
    const bf16_t* __restrict__ A2, long a2bs, int lda2,
    const bf16_t* __restrict__ B2T, long b2bs, int ldb2,
    const float* __restrict__ bias,
    bf16_t* __restrict__ outR, long orbs, int ldoR,
    bf16_t* __restrict__ outT, long otbs, int ldoT, int otcol,
    int apply_gelu) {
    __shared__ __align__(16) bf16_t smem[2][2][ROWS_ * LROW_];  // [buf][matrix]
    int tid = threadIdx.x;
    int wave = tid >> 5, lane = tid & 31;
    int l15 = lane & 15, lh = lane >> 4;
    int b = blockIdx.z;
    int mbase = wave * 16;            // 8 waves cover M = 128
    int nbase = blockIdx.x * ROWS_;
    const bf16_t* b1t = B1T + b * b1bs;
    const bf16_t* b2t = (B2T != nullptr) ? (B2T + b * b2bs) : nullptr;
    const bf16_t* a1row = A1 + b * a1bs + (long)(mbase + l15) * lda1;
    const bf16_t* a2row = (A2 != nullptr) ? (A2 + b * a2bs + (long)(mbase + l15) * lda2)
                                          : nullptr;

    auto stage = [&](int buf, int kc) {   // copy 64 rows x 64 K of B^T per matrix
#pragma unroll
        for (int i = 0; i < 2; ++i) {
            int q = tid + (i << 8);       // 512 x b128 = 8KB per matrix
            int row = q >> 3, c8 = q & 7; // 8 x 16B per row
            copy16_to_lds(&smem[buf][0][row * LROW_ + c8 * 8],
                          b1t + (long)(nbase + row) * ldb1 + kc + c8 * 8);
        }
        if (b2t) {
#pragma unroll
            for (int i = 0; i < 2; ++i) {
                int q = tid + (i << 8);
                int row = q >> 3, c8 = q & 7;
                copy16_to_lds(&smem[buf][1][row * LROW_ + c8 * 8],
                              b2t + (long)(nbase + row) * ldb2 + kc + c8 * 8);
            }
        }
    };

    v8f acc[4] = {};
    int nchunks = K1 / CK_;
    stage(0, 0);
    stage_wait();
    for (int ch = 0; ch < nchunks; ++ch) {
        int cur = ch & 1;
        if (ch + 1 < nchunks) stage(cur ^ 1, (ch + 1) * CK_);   // prefetch next chunk
#pragma unroll
        for (int ks = 0; ks < CK_; ks += 32) {
            int k = ch * CK_ + ks;
            v16bf a1f = load_frag_a(a1row + k, lh);
#pragma unroll
            for (int j = 0; j < 4; ++j) {
                v16bf bb = load_frag_b(&smem[cur][0][(j * 16 + l15) * LROW_ + ks + lh * 16]);
                acc[j] = __builtin_amdgcn_wmma_f32_16x16x32_bf16(false, a1f, false, bb,
                                                                 (short)0, acc[j], false, false);
            }
            if (a2row) {
                v16bf a2f = load_frag_a(a2row + k, lh);
#pragma unroll
                for (int j = 0; j < 4; ++j) {
                    v16bf bb = load_frag_b(&smem[cur][1][(j * 16 + l15) * LROW_ + ks + lh * 16]);
                    acc[j] = __builtin_amdgcn_wmma_f32_16x16x32_bf16(false, a2f, false, bb,
                                                                     (short)0, acc[j], false, false);
                }
            }
        }
        if (ch + 1 < nchunks) stage_wait();
    }
#pragma unroll
    for (int j = 0; j < 4; ++j) {
        int ncol = nbase + j * 16 + l15;
#pragma unroll
        for (int r = 0; r < 8; ++r) {
            int m = mbase + r + 8 * lh;
            float v = acc[j][r];
            if (bias) v += bias[m];
            if (apply_gelu) v = gelu_exact(v);
            bf16_t bv = (bf16_t)v;
            if (outR) outR[b * orbs + (long)m * ldoR + ncol] = bv;
            if (outT) outT[b * otbs + (long)ncol * ldoT + otcol + m] = bv;
        }
    }
}

// ---- fc2: per-node dot over 128 features
__global__ void fc2_kernel(const bf16_t* __restrict__ h1T, // [B][N][128]
                           const float* __restrict__ fc2_w,
                           const float* __restrict__ fc2_b,
                           float* __restrict__ out) {
    int t = blockIdx.x * blockDim.x + threadIdx.x;
    if (t >= B_ * N_) return;
    const bf16_t* h = h1T + (long)t * FC_;
    float acc = fc2_b[0];
#pragma unroll 8
    for (int f = 0; f < FC_; ++f) acc += (float)h[f] * fc2_w[f];
    out[t] = acc;
}

extern "C" void kernel_launch(void* const* d_in, const int* in_sizes, int n_in,
                              void* d_out, int out_size, void* d_ws, size_t ws_size,
                              hipStream_t stream) {
    (void)in_sizes; (void)n_in; (void)out_size; (void)ws_size;
    const float* x            = (const float*)d_in[0];
    const float* nodes        = (const float*)d_in[1];
    const float* node_mask    = (const float*)d_in[2];
    const float* node_weights = (const float*)d_in[3];
    const int*   edges        = (const int*)  d_in[4];
    const float* egw          = (const float*)d_in[5];
    const float* modes        = (const float*)d_in[6];
    const float* spL          = (const float*)d_in[7];
    const float* wC           = (const float*)d_in[8];
    const float* wS           = (const float*)d_in[9];
    // d_in[10] = weights_0 is multiplied by an all-zero basis (b0 = zeros) -> unused
    const float* fc0_w = (const float*)d_in[11];
    const float* fc0_b = (const float*)d_in[12];
    const float* ws_w  = (const float*)d_in[13];
    const float* ws_b  = (const float*)d_in[14];
    const float* gws_w = (const float*)d_in[15];
    const float* gws_b = (const float*)d_in[16];
    const float* fc1_w = (const float*)d_in[17];
    const float* fc1_b = (const float*)d_in[18];
    const float* fc2_w = (const float*)d_in[19];
    const float* fc2_b = (const float*)d_in[20];
    float* out = (float*)d_out;

    char* p = (char*)d_ws;
    auto carve = [&](size_t bytes) -> void* {
        void* r = (void*)p;
        p += (bytes + 255) & ~((size_t)255);
        return r;
    };
    bf16_t* BasNT  = (bf16_t*)carve((size_t)B_ * N_ * K2_ * 2);   // [B][N][2K]
    bf16_t* WBTw   = (bf16_t*)carve((size_t)B_ * K2_ * N_ * 2);   // [B][2K][N]
    bf16_t* hcat0  = (bf16_t*)carve((size_t)B_ * N_ * K2_ * 2);   // [B][N][4C] = hT|gradT
    bf16_t* hcat1  = (bf16_t*)carve((size_t)B_ * N_ * K2_ * 2);
    bf16_t* h_bf   = (bf16_t*)carve((size_t)B_ * C_ * N_ * 2);    // [B][C][N]
    bf16_t* h1T    = (bf16_t*)carve((size_t)B_ * N_ * FC_ * 2);
    float*  gradT  = (float*) carve((size_t)B_ * N_ * GD_ * 4);
    float*  Xf     = (float*) carve((size_t)B_ * C_ * K2_ * 4);
    bf16_t* G      = (bf16_t*)carve((size_t)B_ * C_ * K2_ * 2);
    bf16_t* Wcat   = (bf16_t*)carve((size_t)C_ * K2_ * 2);
    float*  biasCat= (float*) carve((size_t)C_ * 4);
    bf16_t* fc1wb  = (bf16_t*)carve((size_t)FC_ * C_ * 2);

    bases_kernel<<<(B_ * N_ * K_) / 256, 256, 0, stream>>>(nodes, node_mask, node_weights,
                                                           modes, spL, BasNT, WBTw);
    lift_kernel<<<B_ * N_, C_, 0, stream>>>(x, fc0_w, fc0_b, h_bf, hcat0);
    convert_kernel<<<64, 256, 0, stream>>>(fc1_w, fc1wb, (long)FC_ * C_);

    bf16_t* hc[2] = {hcat0, hcat1};
    int cur = 0;
    for (int l = 0; l < L_; ++l) {
        zero_kernel<<<4096, 256, 0, stream>>>(gradT, (long)B_ * N_ * GD_);
        edge_grad_kernel<<<B_ * E_, C_, 0, stream>>>(edges, egw, hc[cur], gradT);
        grad_convert_kernel<<<4096, 256, 0, stream>>>(gradT, hc[cur]);
        zero_kernel<<<256, 256, 0, stream>>>(Xf, (long)B_ * C_ * K2_);
        gemm_fwd_kernel<<<dim3(K2_ / 64, 1, B_ * 8), 256, 0, stream>>>(h_bf, WBTw, Xf);
        modemix_kernel<<<K_, C_, 0, stream>>>(Xf, wC + (size_t)l * C_ * C_ * K_,
                                              wS + (size_t)l * C_ * C_ * K_, G);
        wcat_kernel<<<(C_ * K2_ + 255) / 256, 256, 0, stream>>>(
            ws_w + (size_t)l * C_ * C_, gws_w + (size_t)l * C_ * GD_,
            ws_b + (size_t)l * C_, gws_b + (size_t)l * C_, Wcat, biasCat);
        gemm_dual_kernel<<<dim3(N_ / ROWS_, 1, B_), 256, 0, stream>>>(
            G, (long)C_ * K2_, K2_, K2_,
            BasNT, (long)N_ * K2_, K2_,
            Wcat, 0L, K2_,
            hc[cur], (long)N_ * K2_, K2_,
            biasCat,
            h_bf, (long)C_ * N_, N_,
            hc[1 - cur], (long)N_ * K2_, K2_, 0,
            (l < L_ - 1) ? 1 : 0);
        cur ^= 1;
    }
    // fc1: gelu(hT x fc1_w^T + b) -> h1T
    gemm_dual_kernel<<<dim3(N_ / ROWS_, 1, B_), 256, 0, stream>>>(
        fc1wb, 0L, C_, C_,
        hc[cur], (long)N_ * K2_, K2_,
        nullptr, 0L, 0,
        nullptr, 0L, 0,
        fc1_b,
        nullptr, 0L, 0,
        h1T, (long)N_ * FC_, FC_, 0,
        1);
    fc2_kernel<<<(B_ * N_ + 255) / 256, 256, 0, stream>>>(h1T, fc2_w, fc2_b, out);
}
